// ACBlock_51032801411575
// MI455X (gfx1250) — matrix operand order repaired and checked by reading, not compiled, not run on hardware
//
#include <hip/hip_runtime.h>

// ---------------------------------------------------------------- types ----
typedef __bf16 bf16;
typedef __attribute__((ext_vector_type(16))) __bf16 v16bf;
typedef __attribute__((ext_vector_type(8)))  __bf16 v8bf;
typedef __attribute__((ext_vector_type(8)))  float  v8f;
typedef __attribute__((ext_vector_type(4)))  unsigned int u32x4;
typedef __attribute__((ext_vector_type(8)))  int i32x8;
typedef __attribute__((ext_vector_type(4)))  int i32x4;

#define DEV __device__ __forceinline__

#if defined(__has_builtin)
# if __has_builtin(__builtin_amdgcn_tensor_load_to_lds)
#  define HAVE_TDM 1
# endif
#endif
#if __has_include(<hip/amd_detail/amd_gfx1250_TDM.h>)
# define TDM_ARGS6 1
#endif

#if HAVE_TDM
# ifdef TDM_ARGS6
#  warning "CDNA5 probe: using 6-arg __builtin_amdgcn_tensor_load_to_lds (TDM header present)"
# else
#  warning "CDNA5 probe: using 5-arg __builtin_amdgcn_tensor_load_to_lds"
# endif
#else
# warning "CDNA5 probe: tensor_load_to_lds builtin NOT available - cooperative fallback"
#endif

// geometry constants
constexpr int  NB  = 4;
constexpr int  C0  = 128;
constexpr int  HH  = 256;
constexpr int  WW  = 256;
constexpr long HW  = (long)HH * WW;        // 65536
constexpr long NCHW = (long)NB * C0 * HW;  // 33554432

DEV v8f wmma_bf16(v16bf a, v16bf b, v8f c) {
  return __builtin_amdgcn_wmma_f32_16x16x32_bf16(false, a, false, b,
                                                 (short)0, c, false, false);
}
DEV v16bf cat8(v8bf a, v8bf b) {
  v16bf r;
#pragma unroll
  for (int i = 0; i < 8; ++i) { r[i] = a[i]; r[i + 8] = b[i]; }
  return r;
}

// ------------------------------------------------------------ packers ------
__global__ __launch_bounds__(256) void pack_x_bf16(const float* __restrict__ in,
                                                   bf16* __restrict__ out, long n) {
  long i = (long)blockIdx.x * blockDim.x + threadIdx.x;
  if (i < n) out[i] = (bf16)in[i];
}

// pack OIHW fp32 weights into WMMA 16-bit A-matrix lane layout:
// wp[((cot*khw + k)*cb + b)*512 + lane*16 + t]
__global__ __launch_bounds__(256) void pack_w_bf16(const float* __restrict__ w,
                                                   bf16* __restrict__ wp,
                                                   int cinBlocks, int khw, int cin,
                                                   long total) {
  long idx = (long)blockIdx.x * blockDim.x + threadIdx.x;
  if (idx >= total) return;
  int t    = (int)(idx & 15);
  int lane = (int)((idx >> 4) & 31);
  long rest = idx >> 9;
  int cb  = (int)(rest % cinBlocks);
  int k   = (int)((rest / cinBlocks) % khw);
  int cot = (int)(rest / ((long)cinBlocks * khw));
  int m  = lane & 15, hi = lane >> 4;
  int kl = hi * 16 + t;
  int co = cot * 16 + m;
  int ci = cb * 32 + kl;
  wp[idx] = (bf16)w[((long)co * cin + ci) * khw + k];
}

// ------------------------------------------------- depthwise 7x7 conv ------
__global__ __launch_bounds__(256) void dw7x7(const float* __restrict__ x,
                                             const float* __restrict__ w,
                                             const float* __restrict__ bias,
                                             bf16* __restrict__ out) {
  __shared__ float tile[22][22];
  __shared__ float wk[49];
  int bc = blockIdx.x >> 8;            // (b,c)
  int tt = blockIdx.x & 255;
  int th = tt >> 4, tw = tt & 15;
  int b = bc / C0, c = bc % C0;
  const float* src = x + ((long)b * C0 + c) * HW;
  for (int e = threadIdx.x; e < 22 * 22; e += 256) {
    int yy = th * 16 - 3 + e / 22;
    int xx = tw * 16 - 3 + e % 22;
    float v = 0.f;
    if (yy >= 0 && yy < HH && xx >= 0 && xx < WW) v = src[(long)yy * WW + xx];
    tile[e / 22][e % 22] = v;
  }
  if (threadIdx.x < 49) wk[threadIdx.x] = w[c * 49 + threadIdx.x];
  __syncthreads();
  int hh = threadIdx.x >> 4, ww = threadIdx.x & 15;
  float s = bias[c];
#pragma unroll
  for (int i = 0; i < 7; ++i)
#pragma unroll
    for (int j = 0; j < 7; ++j)
      s += tile[hh + i][ww + j] * wk[i * 7 + j];
  out[((long)b * C0 + c) * HW + (long)(th * 16 + hh) * WW + tw * 16 + ww] = (bf16)s;
}

// ----------------------------------------- implicit-GEMM 3x3 conv ----------
// Cout = 128 always. One WG = 8 waves: 128(M) x 64(N) output tile.
// Double-buffered LDS stage, one barrier per K-step, vectorized global loads.
template <int CIN>
__global__ __launch_bounds__(256) void conv3x3_wmma(
    const bf16*  __restrict__ in,     // [B][CIN][H][W]
    const bf16*  __restrict__ wp,     // packed A-layout weights
    const float* __restrict__ bias,   // [128]
    bf16*  __restrict__ out_bf,       // optional
    float* __restrict__ out_f32,      // optional
    const float* __restrict__ res1,   // optional [B][128][H][W]
    const float* __restrict__ res2,   // optional [B][128][H][W]
    int relu, int oc_tot, int oc_off) {
  constexpr int CB = CIN / 32;
  constexpr int NS = 3 * CB;           // total K-steps
  __shared__ bf16 Xs[2][68][32];       // double-buffered [w-halo][ci] stage

  int n0 = blockIdx.x * 64;
  int w0 = n0 & 255;
  int h  = (n0 >> 8) & 255;
  int b  = n0 >> 16;
  int wave = threadIdx.x >> 5;
  int lane = threadIdx.x & 31;
  int m = lane & 15, hi = lane >> 4;

  // staging thread mapping
  int tci = threadIdx.x >> 3;          // interior: ci 0..31
  int twc = threadIdx.x & 7;           // interior: 8-pixel chunk 0..7
  int hci = threadIdx.x >> 1;          // halo: ci (threads < 64)
  int hside = threadIdx.x & 1;         // halo: left/right

  auto stage_load = [&](int s, v8bf& iv, bf16& hv) {
    int kh = s / CB, cb = s % CB;
    int hr = h + kh - 1;
    bool hok = (hr >= 0) && (hr < HH);
    const bf16* src = in + ((long)b * CIN + cb * 32) * HW + (long)hr * WW;
#pragma unroll
    for (int j = 0; j < 8; ++j) iv[j] = (bf16)0.f;
    if (hok) iv = *(const v8bf*)(src + (long)tci * HW + w0 + twc * 8);
    hv = (bf16)0.f;
    if (threadIdx.x < 64) {
      int wx = hside ? (w0 + 64) : (w0 - 1);
      if (hok && wx >= 0 && wx < WW) hv = src[(long)hci * HW + wx];
    }
  };
  auto stage_store = [&](int buf, v8bf iv, bf16 hv) {
#pragma unroll
    for (int j = 0; j < 8; ++j) Xs[buf][1 + twc * 8 + j][tci] = iv[j];
    if (threadIdx.x < 64) Xs[buf][hside ? 65 : 0][hci] = hv;
  };

  v8f acc[4] = {};
  v8bf iv; bf16 hv;
  stage_load(0, iv, hv);
  stage_store(0, iv, hv);

  for (int s = 0; s < NS; ++s) {
    __syncthreads();                   // buffer (s&1) ready for all waves
    v8bf ivn; bf16 hvn;
    if (s + 1 < NS) stage_load(s + 1, ivn, hvn);   // overlap with compute
    if (threadIdx.x == 0 && s + 2 < NS) {
      int kh2 = (s + 2) / CB, cb2 = (s + 2) % CB;
      int hr2 = h + kh2 - 1;
      if (hr2 >= 0 && hr2 < HH)
        __builtin_prefetch(in + ((long)b * CIN + cb2 * 32) * HW +
                               (long)hr2 * WW + w0, 0, 3);
    }
    int kh = s / CB, cb = s % CB;
#pragma unroll
    for (int kw = 0; kw < 3; ++kw) {
      int kidx = kh * 3 + kw;
      v16bf af = *(const v16bf*)(wp + ((long)(wave * 9 + kidx) * CB + cb) * 512
                                    + (long)lane * 16);
#pragma unroll
      for (int nt = 0; nt < 4; ++nt) {
        int p = nt * 16 + m;
        v16bf bfr = *(const v16bf*)(&Xs[s & 1][p + kw][hi * 16]);
        acc[nt] = wmma_bf16(af, bfr, acc[nt]);
      }
    }
    if (s + 1 < NS) stage_store((s + 1) & 1, ivn, hvn);
  }

  // epilogue
  float bv[8];
#pragma unroll
  for (int r = 0; r < 8; ++r) bv[r] = bias[wave * 16 + 8 * hi + r];
#pragma unroll
  for (int nt = 0; nt < 4; ++nt) {
    int p = nt * 16 + m;
#pragma unroll
    for (int r = 0; r < 8; ++r) {
      int co = wave * 16 + 8 * hi + r;
      float v = acc[nt][r] + bv[r];
      if (relu) v = fmaxf(v, 0.f);
      long ri = ((long)b * C0 + co) * HW + (long)h * WW + (w0 + p);
      if (res1) v += res1[ri];
      if (res2) v += res2[ri];
      long oi = ((long)b * oc_tot + oc_off + co) * HW + (long)h * WW + (w0 + p);
      if (out_f32) out_f32[oi] = v;
      if (out_bf)  out_bf[oi]  = (bf16)v;
    }
  }
}

// ------------------------------------------- windowed attention ------------
// One WG (8 waves) per 16x16 window. LDS plan (320 KB total):
//  [0,64K)    xs  [c][p]  window input           (TDM target)
//  [64,128K)  qs  [p][c]  q ; reused as xat [q][c]
//  [128,192K) ks  [p][c]  k
//  [192,256K) xpc [p][c]  transposed input (dead after phase1)
//  [192,320K) att [q][p]  softmax probs (overlays xpc)
__global__ __launch_bounds__(256) void attn_win(
    const bf16*  __restrict__ xa,   // [B][128][H][W] (depthwise output)
    const bf16*  __restrict__ wq, const float* __restrict__ bq,
    const bf16*  __restrict__ wk, const float* __restrict__ bk,
    const bf16*  __restrict__ w4, const float* __restrict__ b4,
    bf16* __restrict__ out)         // [B][256][H][W], channels 128..255
{
  extern __shared__ v16bf smem_v[];
  bf16* smem = (bf16*)smem_v;
  bf16* xs  = smem;                 // 32768 el
  bf16* qs  = smem + 32768;         // 32768 el
  bf16* ks  = smem + 65536;         // 32768 el
  bf16* xpc = smem + 98304;         // 32768 el
  bf16* att = smem + 98304;         // 65536 el (overlays xpc)
  bf16* xat = qs;

  const float SCALE = 0.088388347648318447f;   // 1/sqrt(128)

  int win = blockIdx.x;
  int b  = win >> 8;
  int h0 = ((win >> 4) & 15) * 16;
  int w0 = (win & 15) * 16;
  int wv = threadIdx.x >> 5;
  int lane = threadIdx.x & 31;
  int ln = lane & 15, hi = lane >> 4;

  const bf16* gsrc = xa + ((long)b * C0) * HW + (long)h0 * WW + w0;

  // ---- phase 0: window -> xs [c][p] -------------------------------------
#if HAVE_TDM
  if ((threadIdx.x >> 5) == 0) {
    unsigned long long ga = (unsigned long long)(size_t)gsrc;
    u32x4 g0; i32x8 g1; i32x4 g2; i32x4 g3;
    g0[0] = 1u;                                   // count=1
    g0[1] = 0u;                                   // lds_addr = 0 (xs)
    g0[2] = (unsigned)ga;
    g0[3] = ((unsigned)(ga >> 32) & 0x01FFFFFFu) | 0x80000000u;  // type=2
    g1[0] = 0x10000;                              // data_size = 2B
    g1[1] = (int)(256u << 16);                    // tensor_dim0 = W
    g1[2] = (int)(256u << 16);                    // tensor_dim1 = H
    g1[3] = (int)(16u << 16);                     // tile_dim0 = 16
    g1[4] = (int)(16u | (128u << 16));            // tile_dim1=16, tile_dim2=128
    g1[5] = 256;                                  // dim0 stride = W
    g1[6] = 0;                                    // dim1 stride lo16 (65536)
    g1[7] = 1;                                    // dim1 stride hi
    g2[0] = 128; g2[1] = 0; g2[2] = 0; g2[3] = 0; // tensor_dim2 = C
    g3[0] = 0; g3[1] = 0; g3[2] = 0; g3[3] = 0;
#ifdef TDM_ARGS6
    i32x8 gz = {};
    __builtin_amdgcn_tensor_load_to_lds(g0, g1, g2, g3, gz, 0);
#else
    __builtin_amdgcn_tensor_load_to_lds(g0, g1, g2, g3, 0);
#endif
    __builtin_amdgcn_s_wait_tensorcnt(0);
  }
#else
  for (int i = threadIdx.x; i < 128 * 256; i += 256) {
    int c = i >> 8, p = i & 255;
    xs[i] = gsrc[(long)c * HW + (long)(p >> 4) * WW + (p & 15)];
  }
#endif
  __syncthreads();

  // transpose xs -> xpc [p][c]
  for (int i = threadIdx.x; i < 128 * 32; i += 256) {
    int c = i & 127, p8 = i >> 7;
    v8bf v = *(const v8bf*)(xs + (long)c * 256 + p8 * 8);
#pragma unroll
    for (int j = 0; j < 8; ++j) xpc[(long)(p8 * 8 + j) * 128 + c] = v[j];
  }
  __syncthreads();

  // ---- phase 1: q/k projections -----------------------------------------
  {
    v16bf wqf[4], wkf[4];
#pragma unroll
    for (int cb = 0; cb < 4; ++cb) {
      wqf[cb] = *(const v16bf*)(wq + ((long)(wv * 4 + cb)) * 512 + (long)lane * 16);
      wkf[cb] = *(const v16bf*)(wk + ((long)(wv * 4 + cb)) * 512 + (long)lane * 16);
    }
    float bqv[8], bkv[8];
#pragma unroll
    for (int r = 0; r < 8; ++r) {
      bqv[r] = bq[wv * 16 + 8 * hi + r];
      bkv[r] = bk[wv * 16 + 8 * hi + r];
    }
    for (int nt = 0; nt < 16; ++nt) {
      int p = nt * 16 + ln;
      v8f aq = {}, ak = {};
#pragma unroll
      for (int cb = 0; cb < 4; ++cb) {
        v16bf bfr = *(const v16bf*)(xpc + (long)p * 128 + cb * 32 + hi * 16);
        aq = wmma_bf16(wqf[cb], bfr, aq);
        ak = wmma_bf16(wkf[cb], bfr, ak);
      }
      v8bf pq, pk;
#pragma unroll
      for (int r = 0; r < 8; ++r) {
        pq[r] = (bf16)(aq[r] + bqv[r]);
        pk[r] = (bf16)(ak[r] + bkv[r]);
      }
      *(v8bf*)(qs + (long)p * 128 + wv * 16 + 8 * hi) = pq;
      *(v8bf*)(ks + (long)p * 128 + wv * 16 + 8 * hi) = pk;
    }
  }
  __syncthreads();

  // ---- phase 2: S = k^T q, softmax, store att [q][p] --------------------
  for (int s = 0; s < 2; ++s) {
    int pb = s * 128 + wv * 16;
    v8f acc[16] = {};
    for (int cb = 0; cb < 4; ++cb) {
      const bf16* ap = ks + (long)(pb + ln) * 128 + cb * 32;
      v16bf af = cat8(*(const v8bf*)(ap + hi * 16),
                      *(const v8bf*)(ap + hi * 16 + 8));
#pragma unroll
      for (int nt = 0; nt < 16; ++nt) {
        v16bf bfr = *(const v16bf*)(qs + (long)(nt * 16 + ln) * 128 + cb * 32 + hi * 16);
        acc[nt] = wmma_bf16(af, bfr, acc[nt]);
      }
    }
    float inv[8];
#pragma unroll
    for (int r = 0; r < 8; ++r) {
      float mx = -1e30f;
#pragma unroll
      for (int nt = 0; nt < 16; ++nt) mx = fmaxf(mx, acc[nt][r]);
      for (int d = 1; d < 16; d <<= 1) mx = fmaxf(mx, __shfl_xor(mx, d, 16));
      float sum = 0.f;
#pragma unroll
      for (int nt = 0; nt < 16; ++nt) {
        float e = __expf((acc[nt][r] - mx) * SCALE);
        acc[nt][r] = e;
        sum += e;
      }
      for (int d = 1; d < 16; d <<= 1) sum += __shfl_xor(sum, d, 16);
      inv[r] = 1.f / sum;
    }
#pragma unroll
    for (int nt = 0; nt < 16; ++nt) {
      v8bf pk;
#pragma unroll
      for (int r = 0; r < 8; ++r) pk[r] = (bf16)(acc[nt][r] * inv[r]);
      *(v8bf*)(att + (long)(nt * 16 + ln) * 256 + pb + 8 * hi) = pk;
    }
  }
  __syncthreads();

  // ---- phase 3: xat^T[q][c] = att^T(qxp) * xb^T(pxc) --------------------
  for (int s = 0; s < 2; ++s) {
    int qb = s * 128 + wv * 16;
    v8f a3[8] = {};
    for (int kp = 0; kp < 8; ++kp) {
      const bf16* ap = att + (long)(qb + ln) * 256 + kp * 32;
      v16bf af = cat8(*(const v8bf*)(ap + hi * 16),
                      *(const v8bf*)(ap + hi * 16 + 8));
#pragma unroll
      for (int nt = 0; nt < 8; ++nt) {
        v16bf bfr = *(const v16bf*)(xs + (long)(nt * 16 + ln) * 256 + kp * 32 + hi * 16);
        a3[nt] = wmma_bf16(af, bfr, a3[nt]);
      }
    }
#pragma unroll
    for (int nt = 0; nt < 8; ++nt)
#pragma unroll
      for (int r = 0; r < 8; ++r)
        xat[(long)(qb + 8 * hi + r) * 128 + nt * 16 + ln] = (bf16)a3[nt][r];
  }
  __syncthreads();

  // ---- phase 4: xo = aw4 * xat + b4 -> global (channels 128..255) -------
  {
    v16bf w4f[4];
#pragma unroll
    for (int cb = 0; cb < 4; ++cb)
      w4f[cb] = *(const v16bf*)(w4 + ((long)(wv * 4 + cb)) * 512 + (long)lane * 16);
    float bv[8];
#pragma unroll
    for (int r = 0; r < 8; ++r) bv[r] = b4[wv * 16 + 8 * hi + r];
    for (int nt = 0; nt < 16; ++nt) {
      int q = nt * 16 + ln;
      v8f ao = {};
#pragma unroll
      for (int cb = 0; cb < 4; ++cb) {
        v16bf bfr = *(const v16bf*)(xat + (long)q * 128 + cb * 32 + hi * 16);
        ao = wmma_bf16(w4f[cb], bfr, ao);
      }
      int ph = q >> 4, pw = q & 15;
#pragma unroll
      for (int r = 0; r < 8; ++r) {
        int o = wv * 16 + 8 * hi + r;
        out[((long)b * 256 + 128 + o) * HW + (long)(h0 + ph) * WW + (w0 + pw)] =
            (bf16)(ao[r] + bv[r]);
      }
    }
  }
}

// ------------------------------------------------------------ host ---------
extern "C" void kernel_launch(void* const* d_in, const int* in_sizes, int n_in,
                              void* d_out, int out_size, void* d_ws, size_t ws_size,
                              hipStream_t stream) {
  const float* x    = (const float*)d_in[0];
  const float* r1w1 = (const float*)d_in[1];
  const float* r1b1 = (const float*)d_in[2];
  const float* r1w2 = (const float*)d_in[3];
  const float* r1b2 = (const float*)d_in[4];
  const float* aw1  = (const float*)d_in[5];
  const float* ab1  = (const float*)d_in[6];
  const float* aw2  = (const float*)d_in[7];
  const float* ab2  = (const float*)d_in[8];
  const float* aw3  = (const float*)d_in[9];
  const float* ab3  = (const float*)d_in[10];
  const float* aw4  = (const float*)d_in[11];
  const float* ab4  = (const float*)d_in[12];
  const float* fw   = (const float*)d_in[13];
  const float* fb   = (const float*)d_in[14];
  const float* r2w1 = (const float*)d_in[15];
  const float* r2b1 = (const float*)d_in[16];
  const float* r2w2 = (const float*)d_in[17];
  const float* r2b2 = (const float*)d_in[18];
  float* out = (float*)d_out;

  // workspace carve-up
  bf16* xbf = (bf16*)d_ws;           // NCHW  (reused as fused_bf after conv1)
  bf16* t1  = xbf + NCHW;            // NCHW  (reused as t2)
  bf16* xab = t1 + NCHW;             // NCHW  depthwise output
  bf16* fin = xab + NCHW;            // 2*NCHW concat buffer [B][256][H][W]
  float* ff = (float*)(fin + 2 * NCHW);  // NCHW fp32 fused
  bf16* wp  = (bf16*)(ff + NCHW);
  const long WP_CONV = 8L * 9 * 4 * 512;   // 147456
  const long WP_FUSE = 8L * 9 * 8 * 512;   // 294912
  const long WP_ATT  = 8L * 1 * 4 * 512;   // 16384
  bf16* wp_r1w1 = wp;
  bf16* wp_r1w2 = wp_r1w1 + WP_CONV;
  bf16* wp_f    = wp_r1w2 + WP_CONV;
  bf16* wp_r2w1 = wp_f + WP_FUSE;
  bf16* wp_r2w2 = wp_r2w1 + WP_CONV;
  bf16* wp_a2   = wp_r2w2 + WP_CONV;
  bf16* wp_a3   = wp_a2 + WP_ATT;
  bf16* wp_a4   = wp_a3 + WP_ATT;

  // ---- packing -----------------------------------------------------------
  pack_x_bf16<<<(unsigned)((NCHW + 255) / 256), 256, 0, stream>>>(x, xbf, NCHW);
  pack_w_bf16<<<(unsigned)((WP_CONV + 255) / 256), 256, 0, stream>>>(r1w1, wp_r1w1, 4, 9, 128, WP_CONV);
  pack_w_bf16<<<(unsigned)((WP_CONV + 255) / 256), 256, 0, stream>>>(r1w2, wp_r1w2, 4, 9, 128, WP_CONV);
  pack_w_bf16<<<(unsigned)((WP_FUSE + 255) / 256), 256, 0, stream>>>(fw,   wp_f,    8, 9, 256, WP_FUSE);
  pack_w_bf16<<<(unsigned)((WP_CONV + 255) / 256), 256, 0, stream>>>(r2w1, wp_r2w1, 4, 9, 128, WP_CONV);
  pack_w_bf16<<<(unsigned)((WP_CONV + 255) / 256), 256, 0, stream>>>(r2w2, wp_r2w2, 4, 9, 128, WP_CONV);
  pack_w_bf16<<<(unsigned)((WP_ATT + 255) / 256), 256, 0, stream>>>(aw2, wp_a2, 4, 1, 128, WP_ATT);
  pack_w_bf16<<<(unsigned)((WP_ATT + 255) / 256), 256, 0, stream>>>(aw3, wp_a3, 4, 1, 128, WP_ATT);
  pack_w_bf16<<<(unsigned)((WP_ATT + 255) / 256), 256, 0, stream>>>(aw4, wp_a4, 4, 1, 128, WP_ATT);

  const unsigned convGrid = (unsigned)(NB * HH * WW / 64);   // 4096

  // ---- ResBlock 1 --------------------------------------------------------
  conv3x3_wmma<128><<<convGrid, 256, 0, stream>>>(
      xbf, wp_r1w1, r1b1, t1, nullptr, nullptr, nullptr, 1, 128, 0);
  conv3x3_wmma<128><<<convGrid, 256, 0, stream>>>(
      t1, wp_r1w2, r1b2, fin, nullptr, x, nullptr, 0, 256, 0);  // cnn -> fin[0:128]

  // ---- Attention branch --------------------------------------------------
  dw7x7<<<(unsigned)(NB * C0 * 256), 256, 0, stream>>>(x, aw1, ab1, xab);
  attn_win<<<(unsigned)(NB * 16 * 16), 256, 327680, stream>>>(
      xab, wp_a2, ab2, wp_a3, ab3, wp_a4, ab4, fin);            // -> fin[128:256]

  // ---- fuse + ResBlock 2 + outer residual --------------------------------
  conv3x3_wmma<256><<<convGrid, 256, 0, stream>>>(
      fin, wp_f, fb, xbf /*fused_bf*/, ff, nullptr, nullptr, 0, 128, 0);
  conv3x3_wmma<128><<<convGrid, 256, 0, stream>>>(
      xbf, wp_r2w1, r2b1, t1 /*t2*/, nullptr, nullptr, nullptr, 1, 128, 0);
  conv3x3_wmma<128><<<convGrid, 256, 0, stream>>>(
      t1, wp_r2w2, r2b2, nullptr, out, ff, x, 0, 128, 0);       // out = x+fused+conv
}
